// Encoder_Postnet_66760971649240
// MI455X (gfx1250) — compile-verified, matrix operand order checked by reading, not run on hardware
//
#include <hip/hip_runtime.h>
#include <hip/hip_bf16.h>

#define Bc 16
#define Sc 512
#define Tc 4096
#define Ec 256

typedef __attribute__((ext_vector_type(16))) __bf16 v16bf;
typedef __attribute__((ext_vector_type(8)))  float  v8f;

__device__ __forceinline__ unsigned short f2bf(float f) {
    // round-to-nearest-even fp32 -> bf16
    unsigned int u = __float_as_uint(f);
    u += 0x7FFFu + ((u >> 16) & 1u);
    return (unsigned short)(u >> 16);
}

// ---------------------------------------------------------------------------
// Kernel 1: serial aligner scan, one block per batch.
// idx_t = (a_t == text[idx_{t-1}]) ? idx_{t-1} : min(idx_{t-1}+1, S-1)
// ---------------------------------------------------------------------------
__global__ void align_kernel(const int* __restrict__ align_phone,
                             const int* __restrict__ text_phone,
                             int* __restrict__ idx_out) {
    __shared__ int tp[Sc];
    const int b = blockIdx.x;
    for (int s = threadIdx.x; s < Sc; s += blockDim.x)
        tp[s] = text_phone[b * Sc + s];
    __syncthreads();
    if (threadIdx.x == 0) {
        int cur = 0;
        int curph = tp[0];
        idx_out[b * Tc] = 0;
        for (int t = 1; t < Tc; ++t) {
            int a = align_phone[b * Tc + t];
            if (a != curph) {
                cur = (cur + 1 < Sc - 1) ? cur + 1 : Sc - 1;
                curph = tp[cur];   // phone only reloaded on advance (~S times total)
            }
            idx_out[b * Tc + t] = cur;
        }
    }
}

// ---------------------------------------------------------------------------
// Kernel 2: W_pos fp32 [K=256][N=256] -> bf16 in WMMA B-fragment lane layout.
// Layout: wB[ ((kb*16 + nb)*32 + lane)*16 + j ]
//   lane<16 : n = nb*16+lane,    k = kb*32 + {0..7, 16..23}  (pairs low/high)
//   lane>=16: n = nb*16+lane-16, k = kb*32 + {8..15, 24..31}
//   ko = (j&7) | (hi<<3) | ((j>>3)<<4)
// ---------------------------------------------------------------------------
__global__ void prep_wpos_kernel(const float* __restrict__ W_pos,
                                 unsigned short* __restrict__ wB) {
    const int g    = blockIdx.x * blockDim.x + threadIdx.x;  // 0..4095
    const int lane = g & 31;
    const int tile = g >> 5;          // kb*16 + nb
    const int nb   = tile & 15;
    const int kb   = tile >> 4;
    const int nl   = lane & 15;
    const int hi   = lane >> 4;
    const int n    = nb * 16 + nl;

    unsigned int pk[8];
#pragma unroll
    for (int jj = 0; jj < 8; ++jj) {
        const int j0  = 2 * jj;
        const int ko0 = (j0 & 7) | (hi << 3) | ((j0 >> 3) << 4);
        const int ko1 = ((j0 + 1) & 7) | (hi << 3) | (((j0 + 1) >> 3) << 4);
        const unsigned short lo = f2bf(W_pos[(kb * 32 + ko0) * Ec + n]);
        const unsigned short hi16 = f2bf(W_pos[(kb * 32 + ko1) * Ec + n]);
        pk[jj] = (unsigned int)lo | ((unsigned int)hi16 << 16);
    }
    unsigned int* dst = (unsigned int*)&wB[((size_t)tile * 32 + lane) * 16];
#pragma unroll
    for (int jj = 0; jj < 8; ++jj) dst[jj] = pk[jj];
}

// ---------------------------------------------------------------------------
// Kernel 3: fused gather + PE + bf16 WMMA GEMM + epilogue.
// One block = 16 rows (one batch, 16 consecutive t) x full N=256.
// 512 threads = 16 waves; wave w computes the 16x16 tile at columns [16w,16w+16).
// ---------------------------------------------------------------------------
__global__ __launch_bounds__(512) void postnet_kernel(
    const float* __restrict__ encoder_out,
    const float* __restrict__ pitch,
    const int*   __restrict__ beats,
    const float* __restrict__ W_pitch,
    const float* __restrict__ b_pitch,
    const float* __restrict__ emb_beats,
    const float* __restrict__ b_pos,
    const int*   __restrict__ idx_in,
    const unsigned short* __restrict__ wB,
    float* __restrict__ out) {

    __shared__ __align__(32) unsigned short ldsA[8 * 32 * 16];  // 8 KB, A-fragment order
    __shared__ float ldsAl[16 * 256];                           // 16 KB, raw aligner rows
    __shared__ float ldsPitch[16];
    __shared__ int   ldsBeat[16];

    const int blk = blockIdx.x;
    const int tilesPerB = Tc / 16;
    const int b  = blk / tilesPerB;
    const int t0 = (blk % tilesPerB) * 16;
    const int tid = threadIdx.x;

    // ---- stage: gather + positional encoding, write fp32 row + swizzled bf16 A
    {
        const int m  = tid >> 5;      // row 0..15
        const int w  = tid & 31;      // 32 threads per row
        const int k0 = w * 8;         // 8 consecutive k per thread
        const int t  = t0 + m;
        const int src = idx_in[b * Tc + t];
        const float* erow = encoder_out + ((size_t)b * Sc + src) * Ec;
        if (w == 0) {
            ldsPitch[m] = pitch[b * Tc + t];
            ldsBeat[m]  = beats[b * Tc + t];
        }
        float vals[8];
#pragma unroll
        for (int q = 0; q < 8; ++q) {
            const int e = k0 + q;
            const float al = erow[e];
            // PE: i = e>>1; freq = exp(-(ln 1e4 / E) * 2i); even->sin, odd->cos
            const float i2   = (float)(e & ~1);
            const float freq = __expf(-9.210340371976184f * (1.0f / 256.0f) * i2);
            const float ang  = (float)t * freq;
            const float pe   = (e & 1) ? __cosf(ang) : __sinf(ang);
            ldsAl[m * 256 + e] = al;
            vals[q] = al + pe;   // GEMM input row
        }
        // bf16 store in A-fragment order: kb=k0>>5, hi=(ko>>3)&1, j=(ko&7)+8*(ko>>4)
        const int kb  = k0 >> 5;
        const int ko  = k0 & 31;
        const int hi  = (ko >> 3) & 1;
        const int ko4 = ko >> 4;
        const int lane_t = m + 16 * hi;
        const int base = (kb * 32 + lane_t) * 16 + 8 * ko4;  // 16B-aligned (ushort idx)
        unsigned int pk[4];
#pragma unroll
        for (int q = 0; q < 4; ++q)
            pk[q] = (unsigned int)f2bf(vals[2 * q]) |
                    ((unsigned int)f2bf(vals[2 * q + 1]) << 16);
        uint4 v; v.x = pk[0]; v.y = pk[1]; v.z = pk[2]; v.w = pk[3];
        *(uint4*)&ldsA[base] = v;
    }
    __syncthreads();

    // ---- GEMM: 8x v_wmma_f32_16x16x32_bf16 per wave (K = 256)
    const int wave = tid >> 5;   // nb: column strip [16*wave, 16*wave+16)
    const int lane = tid & 31;
    v8f acc = {};
#pragma unroll
    for (int kb = 0; kb < 8; ++kb) {
        __builtin_prefetch(&wB[(((kb + 1) & 7) * 16 + wave) * 32 * 16], 0, 1);
        v16bf afrag = *(const v16bf*)&ldsA[(kb * 32 + lane) * 16];
        v16bf bfrag = *(const v16bf*)&wB[(((size_t)kb * 16 + wave) * 32 + lane) * 16];
        acc = __builtin_amdgcn_wmma_f32_16x16x32_bf16(
            /*neg_a=*/false, afrag, /*neg_b=*/false, bfrag,
            /*c_mod=*/(short)0, acc, /*reuse_a=*/false, /*reuse_b=*/false);
    }

    // ---- epilogue: C/D layout -> lane(l<16): N=l, M=v ; lane>=16: N=l-16, M=v+8
    const int nl = lane & 15;
    const int hi = lane >> 4;
    const int n  = wave * 16 + nl;
    const float wn   = W_pitch[n];
    const float bp   = b_pitch[n];
    const float bpos = b_pos[n];
    const float eb0  = emb_beats[n];
    const float eb1  = emb_beats[Ec + n];
#pragma unroll
    for (int v = 0; v < 8; ++v) {
        const int m = v + 8 * hi;
        const int t = t0 + m;
        const float beatv = ldsBeat[m] ? eb1 : eb0;
        const float val = acc[v] + bpos + ldsAl[m * 256 + n] +
                          ldsPitch[m] * wn + bp + beatv;
        out[((size_t)b * Tc + t) * Ec + n] = val;
    }
}

// ---------------------------------------------------------------------------
extern "C" void kernel_launch(void* const* d_in, const int* in_sizes, int n_in,
                              void* d_out, int out_size, void* d_ws, size_t ws_size,
                              hipStream_t stream) {
    const float* encoder_out = (const float*)d_in[0];
    const int*   align_phone = (const int*)d_in[1];
    const int*   text_phone  = (const int*)d_in[2];
    const float* pitch       = (const float*)d_in[3];
    const int*   beats       = (const int*)d_in[4];
    const float* W_pitch     = (const float*)d_in[5];
    const float* b_pitch     = (const float*)d_in[6];
    const float* emb_beats   = (const float*)d_in[7];
    const float* W_pos       = (const float*)d_in[8];
    const float* b_pos       = (const float*)d_in[9];
    float* out = (float*)d_out;

    int* idxbuf = (int*)d_ws;                                   // B*T*4  = 256 KB
    unsigned short* wB =
        (unsigned short*)((char*)d_ws + (size_t)Bc * Tc * 4);   // 128 KB bf16

    align_kernel<<<Bc, 128, 0, stream>>>(align_phone, text_phone, idxbuf);
    prep_wpos_kernel<<<4096 / 256, 256, 0, stream>>>(W_pos, wB);
    postnet_kernel<<<Bc * Tc / 16, 512, 0, stream>>>(
        encoder_out, pitch, beats, W_pitch, b_pitch, emb_beats, b_pos,
        idxbuf, wB, out);
}
